// AFNOMixingBlock_11527692222558
// MI455X (gfx1250) — matrix operand order repaired and checked by reading, not compiled, not run on hardware
//
#include <hip/hip_runtime.h>
#include <math.h>

typedef __attribute__((ext_vector_type(16))) _Float16 v16h;
typedef __attribute__((ext_vector_type(8)))  _Float16 v8h;
typedef __attribute__((ext_vector_type(8)))  float    v8f;

#define Bn   4
#define Hn   128
#define Wn   128
#define Cn   768
#define WF   65
#define WFP  96
#define NBLK 8
#define BS   96
#define LAM  0.01f

// ---------------------------------------------------------------------------
// WMMA helpers (gfx1250: v_wmma_f32_16x16x32_f16, wave32)
// ---------------------------------------------------------------------------
__device__ __forceinline__ v8f wmma_f16(v16h a, v16h b, v8f c) {
  return __builtin_amdgcn_wmma_f32_16x16x32_f16(false, a, false, b, (short)0, c,
                                                false, false);
}

// A-fragment (16x32 f16, MxK): lane l holds row (l&15); halves 0-7 at K offset
// (l>>4)*8, halves 8-15 at +16.  `p` already points at (row, kbase) element.
__device__ __forceinline__ v16h load_afrag(const _Float16* p) {
  v8h lo = *(const v8h*)(p);
  v8h hi = *(const v8h*)(p + 16);
  return __builtin_shufflevector(lo, hi, 0, 1, 2, 3, 4, 5, 6, 7,
                                 8, 9, 10, 11, 12, 13, 14, 15);
}

// B-fragment (32x16 f16, KxN), pre-swizzled in memory: frag f -> 32 lanes x 16
// contiguous halves.
__device__ __forceinline__ v16h load_bfrag(const _Float16* tab, int frag, int lane) {
  return *(const v16h*)(tab + ((size_t)frag * 32 + lane) * 16);
}

__device__ __forceinline__ float sshrink(float x) {
  float t = fmaxf(fabsf(x) - LAM, 0.0f);
  return copysignf(t, x);
}

// ---------------------------------------------------------------------------
// Twiddle table generator: writes B-fragments (KxN) of
//   mode 0:  scale * [ck] * cos(2*pi*k*n/128)
//   mode 1:  scale * [ck] * sign * sin(2*pi*k*n/128)
// zeroed for k >= Kvalid or n >= Nvalid.
// ---------------------------------------------------------------------------
__global__ void k_twiddle(_Float16* __restrict__ dst, int F, int NT, int Kvalid,
                          int Nvalid, float scale, float sign, int mode, int useCk) {
  int idx = blockIdx.x * 256 + threadIdx.x;
  int lane = idx & 31, f = idx >> 5;
  if (f >= F) return;
  int kt = f / NT, nt = f % NT;
  int n  = nt * 16 + (lane & 15);
  int kb = kt * 32 + (lane >> 4) * 8;
  for (int h = 0; h < 16; h++) {
    int k = kb + (h < 8 ? h : h + 8);
    float v = 0.0f;
    if (k < Kvalid && n < Nvalid) {
      int ph = (k * n) & 127;
      float th = 6.283185307179586f * (float)ph / 128.0f;
      v = (mode == 0) ? cosf(th) : sign * sinf(th);
      if (useCk) v *= (k == 0 || k == 64) ? 1.0f : 2.0f;
      v *= scale;
    }
    dst[(size_t)f * 512 + lane * 16 + h] = (_Float16)v;
  }
}

// Weight B-fragments: mat 0..5 = w1r, w1i, -w1i, w2r, w2i, -w2i per block.
__global__ void k_wfrag(const float* __restrict__ w1, const float* __restrict__ w2,
                        _Float16* __restrict__ dst) {
  int idx = blockIdx.x * 256 + threadIdx.x;
  int lane = idx & 31, f = idx >> 5;
  if (f >= NBLK * 6 * 18) return;
  int blk = f / (6 * 18);
  int rem = f % (6 * 18);
  int mat = rem / 18;
  int fi  = rem % 18;
  int kt = fi / 6, nt = fi % 6;
  const float* w = (mat < 3) ? w1 : w2;
  int ri = (mat % 3 == 0) ? 0 : 1;
  float sgn = (mat % 3 == 2) ? -1.0f : 1.0f;
  int o  = nt * 16 + (lane & 15);
  int kb = kt * 32 + (lane >> 4) * 8;
  for (int h = 0; h < 16; h++) {
    int i = kb + (h < 8 ? h : h + 8);
    float v = sgn * w[((size_t)(ri * NBLK + blk) * BS + i) * BS + o];
    dst[(size_t)f * 512 + lane * 16 + h] = (_Float16)v;
  }
}

// ---------------------------------------------------------------------------
// Generic LDS tile transpose:
//   src[b0*sB0 + b1*sB1 + i*sI + j]  ->  dst[b0*dB0 + b1*dB1 + j*dJ + i]
// MODE 0: f16->f16   MODE 1: f32->f16   MODE 2: f16->f32 + residual add
// ---------------------------------------------------------------------------
template <int MODE>
__global__ __launch_bounds__(256) void k_transpose(
    const void* __restrict__ srcv, void* __restrict__ dstv,
    const float* __restrict__ res, int nB1, int nI, int nJ,
    long long sB0, long long sB1, long long sI,
    long long dB0, long long dB1, long long dJ) {
  __shared__ float tile[32][33];
  int b0 = blockIdx.z / nB1, b1 = blockIdx.z % nB1;
  int i0 = blockIdx.y * 32, j0 = blockIdx.x * 32;
  int tx = threadIdx.x & 31, ty = threadIdx.x >> 5;
  long long sbase = (long long)b0 * sB0 + (long long)b1 * sB1;
  long long dbase = (long long)b0 * dB0 + (long long)b1 * dB1;
#pragma unroll
  for (int r = 0; r < 4; r++) {
    int ii = i0 + ty + r * 8, jj = j0 + tx;
    float v = 0.0f;
    if (ii < nI && jj < nJ) {
      long long o = sbase + (long long)ii * sI + jj;
      if (MODE == 1) v = ((const float*)srcv)[o];
      else           v = (float)((const _Float16*)srcv)[o];
    }
    tile[ty + r * 8][tx] = v;
  }
  __syncthreads();
#pragma unroll
  for (int r = 0; r < 4; r++) {
    int jj = j0 + ty + r * 8, ii = i0 + tx;
    if (jj < nJ && ii < nI) {
      float v = tile[tx][ty + r * 8];
      long long o = dbase + (long long)jj * dJ + ii;
      if (MODE == 2) ((float*)dstv)[o] = v + res[o];
      else           ((_Float16*)dstv)[o] = (_Float16)v;
    }
  }
}

// ---------------------------------------------------------------------------
// Stage 1: real rFFT along W as GEMM.  Per slab (b,h):
//   Dr[c,wf] = sum_w A[c,w]*cos*s,  Di[c,wf] = sum_w A[c,w]*(-sin)*s
// A pitch 128 (w), D pitch 96 (wf, cols >= 65 untouched).
// ---------------------------------------------------------------------------
__global__ __launch_bounds__(256) void k_dft_w_real(
    const _Float16* __restrict__ A, const _Float16* __restrict__ tc,
    const _Float16* __restrict__ ts, _Float16* __restrict__ Dr,
    _Float16* __restrict__ Di) {
  int lane = threadIdx.x & 31, wave = threadIdx.x >> 5;
  int mtile = blockIdx.x * 8 + wave;  // 0..47  (c tiles)
  int slab  = blockIdx.y;             // 0..511 (b*h)
  const _Float16* ab = A + (size_t)slab * (Cn * Wn) +
                       (size_t)(mtile * 16 + (lane & 15)) * Wn + (lane >> 4) * 8;
  v8f z = {};
  v8f accR[5], accI[5];
#pragma unroll
  for (int t = 0; t < 5; t++) { accR[t] = z; accI[t] = z; }
#pragma unroll
  for (int kt = 0; kt < 4; kt++) {
    if (kt < 3) __builtin_prefetch(ab + (kt + 1) * 32, 0, 1);
    v16h a = load_afrag(ab + kt * 32);
#pragma unroll
    for (int nt = 0; nt < 5; nt++) {
      accR[nt] = wmma_f16(a, load_bfrag(tc, kt * 5 + nt, lane), accR[nt]);
      accI[nt] = wmma_f16(a, load_bfrag(ts, kt * 5 + nt, lane), accI[nt]);
    }
  }
  int n0 = lane & 15, hw = lane >> 4;
#pragma unroll
  for (int nt = 0; nt < 5; nt++) {
    int n = nt * 16 + n0;
    if (n < WF) {
      size_t o = (size_t)slab * (Cn * WFP) + (size_t)(mtile * 16 + hw * 8) * WFP + n;
#pragma unroll
      for (int v = 0; v < 8; v++) {
        Dr[o + (size_t)v * WFP] = (_Float16)accR[nt][v];
        Di[o + (size_t)v * WFP] = (_Float16)accI[nt][v];
      }
    }
  }
}

// ---------------------------------------------------------------------------
// Complex 128-pt DFT along the K axis (used fwd for H and inverse for H).
//   Dr = Ar*tc + Ai*tsn ,  Di = Ar*ts + Ai*tc     (pitch 128 both sides)
// ---------------------------------------------------------------------------
__global__ __launch_bounds__(256) void k_dft_cplx(
    const _Float16* __restrict__ Ar, const _Float16* __restrict__ Ai,
    const _Float16* __restrict__ tc, const _Float16* __restrict__ ts,
    const _Float16* __restrict__ tsn, _Float16* __restrict__ Dr,
    _Float16* __restrict__ Di) {
  int lane = threadIdx.x & 31, wave = threadIdx.x >> 5;
  int mtile = blockIdx.x * 8 + wave;  // c tile
  int nh    = blockIdx.y;             // 0..1 (n halves)
  int slab  = blockIdx.z;             // 0..259 (b*wf)
  size_t ab = (size_t)slab * (Cn * 128) +
              (size_t)(mtile * 16 + (lane & 15)) * 128 + (lane >> 4) * 8;
  v8f z = {};
  v8f accR[4], accI[4];
#pragma unroll
  for (int t = 0; t < 4; t++) { accR[t] = z; accI[t] = z; }
#pragma unroll
  for (int kt = 0; kt < 4; kt++) {
    if (kt < 3) __builtin_prefetch(Ar + ab + (kt + 1) * 32, 0, 1);
    v16h ar = load_afrag(Ar + ab + kt * 32);
    v16h ai = load_afrag(Ai + ab + kt * 32);
#pragma unroll
    for (int t = 0; t < 4; t++) {
      int f = kt * 8 + nh * 4 + t;
      v16h bc = load_bfrag(tc, f, lane);
      accR[t] = wmma_f16(ar, bc, accR[t]);
      accR[t] = wmma_f16(ai, load_bfrag(tsn, f, lane), accR[t]);
      accI[t] = wmma_f16(ar, load_bfrag(ts, f, lane), accI[t]);
      accI[t] = wmma_f16(ai, bc, accI[t]);
    }
  }
  int n0 = lane & 15, hw = lane >> 4;
#pragma unroll
  for (int t = 0; t < 4; t++) {
    int n = (nh * 4 + t) * 16 + n0;
    size_t o = (size_t)slab * (Cn * 128) + (size_t)(mtile * 16 + hw * 8) * 128 + n;
#pragma unroll
    for (int v = 0; v < 8; v++) {
      Dr[o + (size_t)v * 128] = (_Float16)accR[t][v];
      Di[o + (size_t)v * 128] = (_Float16)accI[t][v];
    }
  }
}

// ---------------------------------------------------------------------------
// Fused block-diagonal complex 2-layer MLP with softshrink.
// P[p][c] (c contiguous, K axis) -> Q[p][c].  One wave per 16-point tile,
// layer-1 outputs staged through LDS to become layer-2 A-fragments.
// ---------------------------------------------------------------------------
__global__ __launch_bounds__(256) void k_mlp(
    const _Float16* __restrict__ Pr, const _Float16* __restrict__ Pi,
    const _Float16* __restrict__ wtab, const float* __restrict__ b1,
    const float* __restrict__ b2, _Float16* __restrict__ Qr,
    _Float16* __restrict__ Qi) {
  __shared__ _Float16 lsr[8][16 * 104];
  __shared__ _Float16 lsi[8][16 * 104];
  int lane = threadIdx.x & 31, wave = threadIdx.x >> 5;
  int ptile = blockIdx.x * 8 + wave;  // 0..2079
  int blk   = blockIdx.y;             // 0..7
  int n0 = lane & 15, hw = lane >> 4;
  size_t ab = (size_t)(ptile * 16 + n0) * Cn + blk * BS + hw * 8;
  const _Float16* wt = wtab + (size_t)blk * (6 * 18 * 512);
  v8f z = {};
  v8f oR[6], oI[6];
#pragma unroll
  for (int t = 0; t < 6; t++) { oR[t] = z; oI[t] = z; }
#pragma unroll
  for (int kt = 0; kt < 3; kt++) {
    v16h xr = load_afrag(Pr + ab + kt * 32);
    v16h xi = load_afrag(Pi + ab + kt * 32);
#pragma unroll
    for (int nt = 0; nt < 6; nt++) {
      int fi = kt * 6 + nt;
      v16h w1r = load_bfrag(wt, 0 * 18 + fi, lane);
      oR[nt] = wmma_f16(xr, w1r, oR[nt]);
      oR[nt] = wmma_f16(xi, load_bfrag(wt, 2 * 18 + fi, lane), oR[nt]);
      oI[nt] = wmma_f16(xr, load_bfrag(wt, 1 * 18 + fi, lane), oI[nt]);
      oI[nt] = wmma_f16(xi, w1r, oI[nt]);
    }
  }
#pragma unroll
  for (int nt = 0; nt < 6; nt++) {
    int o = nt * 16 + n0;
    float br = b1[(0 * NBLK + blk) * BS + o];
    float bi = b1[(1 * NBLK + blk) * BS + o];
#pragma unroll
    for (int v = 0; v < 8; v++) {
      int m = v + 8 * hw;
      lsr[wave][m * 104 + o] = (_Float16)sshrink(oR[nt][v] + br);
      lsi[wave][m * 104 + o] = (_Float16)sshrink(oI[nt][v] + bi);
    }
  }
  __syncthreads();
#pragma unroll
  for (int t = 0; t < 6; t++) { oR[t] = z; oI[t] = z; }
#pragma unroll
  for (int kt = 0; kt < 3; kt++) {
    v16h ar = load_afrag(&lsr[wave][n0 * 104 + kt * 32 + hw * 8]);
    v16h ai = load_afrag(&lsi[wave][n0 * 104 + kt * 32 + hw * 8]);
#pragma unroll
    for (int nt = 0; nt < 6; nt++) {
      int fi = kt * 6 + nt;
      v16h w2r = load_bfrag(wt, 3 * 18 + fi, lane);
      oR[nt] = wmma_f16(ar, w2r, oR[nt]);
      oR[nt] = wmma_f16(ai, load_bfrag(wt, 5 * 18 + fi, lane), oR[nt]);
      oI[nt] = wmma_f16(ar, load_bfrag(wt, 4 * 18 + fi, lane), oI[nt]);
      oI[nt] = wmma_f16(ai, w2r, oI[nt]);
    }
  }
#pragma unroll
  for (int nt = 0; nt < 6; nt++) {
    int o = nt * 16 + n0;
    float br = b2[(0 * NBLK + blk) * BS + o];
    float bi = b2[(1 * NBLK + blk) * BS + o];
    size_t ob = (size_t)(ptile * 16 + 8 * hw) * Cn + blk * BS + o;
#pragma unroll
    for (int v = 0; v < 8; v++) {
      Qr[ob + (size_t)v * Cn] = (_Float16)sshrink(oR[nt][v] + br);
      Qi[ob + (size_t)v * Cn] = (_Float16)sshrink(oI[nt][v] + bi);
    }
  }
}

// ---------------------------------------------------------------------------
// Inverse real synthesis along W:  R[c,w] = sum_wf Zr*tc + Zi*ts
// (tables carry c_k weights, 1/128 ortho scale, and zero rows k>=65).
// ---------------------------------------------------------------------------
__global__ __launch_bounds__(256) void k_idft_w(
    const _Float16* __restrict__ Zr, const _Float16* __restrict__ Zi,
    const _Float16* __restrict__ tc, const _Float16* __restrict__ ts,
    _Float16* __restrict__ R) {
  int lane = threadIdx.x & 31, wave = threadIdx.x >> 5;
  int mtile = blockIdx.x * 8 + wave;
  int slab  = blockIdx.y;  // 0..511 (b*h)
  size_t ab = (size_t)slab * (Cn * WFP) +
              (size_t)(mtile * 16 + (lane & 15)) * WFP + (lane >> 4) * 8;
  v8f z = {};
  v8f acc[8];
#pragma unroll
  for (int t = 0; t < 8; t++) acc[t] = z;
#pragma unroll
  for (int kt = 0; kt < 3; kt++) {
    v16h zr = load_afrag(Zr + ab + kt * 32);
    v16h zi = load_afrag(Zi + ab + kt * 32);
#pragma unroll
    for (int nt = 0; nt < 8; nt++) {
      int f = kt * 8 + nt;
      acc[nt] = wmma_f16(zr, load_bfrag(tc, f, lane), acc[nt]);
      acc[nt] = wmma_f16(zi, load_bfrag(ts, f, lane), acc[nt]);
    }
  }
  int n0 = lane & 15, hw = lane >> 4;
#pragma unroll
  for (int nt = 0; nt < 8; nt++) {
    size_t o = (size_t)slab * (Cn * 128) +
               (size_t)(mtile * 16 + hw * 8) * 128 + nt * 16 + n0;
#pragma unroll
    for (int v = 0; v < 8; v++) R[o + (size_t)v * 128] = (_Float16)acc[nt][v];
  }
}

// ---------------------------------------------------------------------------
// Host-side orchestration
// ---------------------------------------------------------------------------
extern "C" void kernel_launch(void* const* d_in, const int* in_sizes, int n_in,
                              void* d_out, int out_size, void* d_ws, size_t ws_size,
                              hipStream_t stream) {
  const float* x  = (const float*)d_in[0];
  const float* w1 = (const float*)d_in[1];
  const float* w2 = (const float*)d_in[2];
  const float* b1 = (const float*)d_in[3];
  const float* b2 = (const float*)d_in[4];
  float* out = (float*)d_out;
  char* ws = (char*)d_ws;

  constexpr size_t E_A2 = (size_t)512 * 768 * 96;   // pitch-96 half buffer elems
  constexpr size_t E_C  = (size_t)260 * 768 * 128;  // pitch-128 half buffer elems
  constexpr size_t E_A1 = (size_t)512 * 768 * 128;
  constexpr size_t OFF_P1 = 0;
  constexpr size_t OFF_P2 = OFF_P1 + 2 * E_A2 * 2;
  constexpr size_t OFF_P3 = OFF_P2 + 2 * E_C * 2;
  constexpr size_t OFF_P4 = OFF_P3 + 2 * E_C * 2;
  constexpr size_t OFF_T  = OFF_P4 + E_A1 * 2;

  _Float16* A1  = (_Float16*)(ws + OFF_P4);
  _Float16* A2r = (_Float16*)(ws + OFF_P1); _Float16* A2i = A2r + E_A2;
  _Float16* A3r = (_Float16*)(ws + OFF_P2); _Float16* A3i = A3r + E_C;
  _Float16* A4r = (_Float16*)(ws + OFF_P3); _Float16* A4i = A4r + E_C;
  _Float16* Pr  = (_Float16*)(ws + OFF_P1); _Float16* Pi  = Pr + E_C;
  _Float16* Qr  = (_Float16*)(ws + OFF_P2); _Float16* Qi  = Qr + E_C;
  _Float16* A5r = (_Float16*)(ws + OFF_P3); _Float16* A5i = A5r + E_C;
  _Float16* A7r = (_Float16*)(ws + OFF_P2); _Float16* A7i = A7r + E_C;
  _Float16* A6r = (_Float16*)(ws + OFF_P1); _Float16* A6i = A6r + E_A2;
  _Float16* Rb  = (_Float16*)(ws + OFF_P4);

  _Float16* tS1c = (_Float16*)(ws + OFF_T + 0);
  _Float16* tS1s = (_Float16*)(ws + OFF_T + 20480);
  _Float16* tFc  = (_Float16*)(ws + OFF_T + 40960);
  _Float16* tFs  = (_Float16*)(ws + OFF_T + 73728);
  _Float16* tFn  = (_Float16*)(ws + OFF_T + 106496);
  _Float16* tIc  = (_Float16*)(ws + OFF_T + 139264);
  _Float16* tIs  = (_Float16*)(ws + OFF_T + 172032);
  _Float16* tIn  = (_Float16*)(ws + OFF_T + 204800);
  _Float16* tWc  = (_Float16*)(ws + OFF_T + 237568);
  _Float16* tWs  = (_Float16*)(ws + OFF_T + 262144);
  _Float16* wtab = (_Float16*)(ws + OFF_T + 286720);

  auto tw = [&](_Float16* dst, int F, int NT, int Kv, int Nv, float sc, float sg,
                int mode, int ck) {
    k_twiddle<<<dim3((F * 32 + 255) / 256), 256, 0, stream>>>(dst, F, NT, Kv, Nv,
                                                              sc, sg, mode, ck);
  };
  const float s = 1.0f / 128.0f;
  tw(tS1c, 20, 5, 128, WF, s, 1.f, 0, 0);   // fwd-W cos /128
  tw(tS1s, 20, 5, 128, WF, s, -1.f, 1, 0);  // fwd-W -sin /128
  tw(tFc, 32, 8, 128, 128, 1.f, 1.f, 0, 0); // fwd-H cos
  tw(tFs, 32, 8, 128, 128, 1.f, -1.f, 1, 0);
  tw(tFn, 32, 8, 128, 128, 1.f, 1.f, 1, 0);
  tw(tIc, 32, 8, 128, 128, 1.f, 1.f, 0, 0); // inv-H cos
  tw(tIs, 32, 8, 128, 128, 1.f, 1.f, 1, 0);
  tw(tIn, 32, 8, 128, 128, 1.f, -1.f, 1, 0);
  tw(tWc, 24, 8, WF, 128, s, 1.f, 0, 1);    // inv-W c_k cos /128 (rows>=65 = 0)
  tw(tWs, 24, 8, WF, 128, s, -1.f, 1, 1);
  k_wfrag<<<dim3((864 * 32 + 255) / 256), 256, 0, stream>>>(w1, w2, wtab);

  auto tr = [&](int mode, const void* src, void* dst, const float* res, int nB0,
                int nB1, int nI, int nJ, long long sB0, long long sB1,
                long long sI, long long dB0, long long dB1, long long dJ) {
    dim3 g((nJ + 31) / 32, (nI + 31) / 32, nB0 * nB1);
    if (mode == 0)
      k_transpose<0><<<g, 256, 0, stream>>>(src, dst, res, nB1, nI, nJ, sB0, sB1,
                                            sI, dB0, dB1, dJ);
    else if (mode == 1)
      k_transpose<1><<<g, 256, 0, stream>>>(src, dst, res, nB1, nI, nJ, sB0, sB1,
                                            sI, dB0, dB1, dJ);
    else
      k_transpose<2><<<g, 256, 0, stream>>>(src, dst, res, nB1, nI, nJ, sB0, sB1,
                                            sI, dB0, dB1, dJ);
  };

  // T0: x[b,h,w,c] f32 -> A1[(b,h)][c][w] f16
  tr(1, x, A1, nullptr, 512, 1, 128, 768, 98304, 0, 768, 98304, 0, 128);
  // G1: rFFT along W
  k_dft_w_real<<<dim3(6, 512), 256, 0, stream>>>(A1, tS1c, tS1s, A2r, A2i);
  // T1: [(b,h)][c][wf] -> [(b,wf)][c][h]
  tr(0, A2r, A3r, nullptr, 4, 768, 128, WF, 9437184, 96, 73728, 6389760, 128, 98304);
  tr(0, A2i, A3i, nullptr, 4, 768, 128, WF, 9437184, 96, 73728, 6389760, 128, 98304);
  // G2 fwd: complex FFT along H
  k_dft_cplx<<<dim3(6, 2, 260), 256, 0, stream>>>(A3r, A3i, tFc, tFs, tFn, A4r, A4i);
  // T2: [(b,wf)][c][hf] -> P[p][c]
  tr(0, A4r, Pr, nullptr, 4, WF, 768, 128, 6389760, 98304, 128, 6389760, 768, 49920);
  tr(0, A4i, Pi, nullptr, 4, WF, 768, 128, 6389760, 98304, 128, 6389760, 768, 49920);
  // MLP (2 complex layers + softshrink)
  k_mlp<<<dim3(260, 8), 256, 0, stream>>>(Pr, Pi, wtab, b1, b2, Qr, Qi);
  // T3: Q[p][c] -> [(b,wf)][c][hf]
  tr(0, Qr, A5r, nullptr, 4, WF, 128, 768, 6389760, 768, 49920, 6389760, 98304, 128);
  tr(0, Qi, A5i, nullptr, 4, WF, 128, 768, 6389760, 768, 49920, 6389760, 98304, 128);
  // G2 inv: inverse FFT along H
  k_dft_cplx<<<dim3(6, 2, 260), 256, 0, stream>>>(A5r, A5i, tIc, tIs, tIn, A7r, A7i);
  // T4: [(b,wf)][c][h] -> [(b,h)][c][wf]
  tr(0, A7r, A6r, nullptr, 4, 768, WF, 128, 6389760, 128, 98304, 9437184, 96, 73728);
  tr(0, A7i, A6i, nullptr, 4, 768, WF, 128, 6389760, 128, 98304, 9437184, 96, 73728);
  // G4: inverse real synthesis along W
  k_idft_w<<<dim3(6, 512), 256, 0, stream>>>(A6r, A6i, tWc, tWs, Rb);
  // T5: [(b,h)][c][w] f16 -> out[b,h,w,c] f32 + residual x
  tr(2, Rb, out, x, 512, 1, 768, 128, 98304, 0, 128, 98304, 0, 768);
}
